// BlockMultiHeadAttention_42880953483361
// MI455X (gfx1250) — matrix operand order repaired and checked by reading, not compiled, not run on hardware
//
#include <hip/hip_runtime.h>
#include <hip/hip_bf16.h>

// ---------------------------------------------------------------------------
// BlockMultiHeadAttention forward for MI455X (gfx1250, wave32, WMMA bf16).
//
// Roofline: 805 MB of fp32 projection weights dominate (~35us at 23.3 TB/s);
// all GEMMs use v_wmma_f32_16x16x32_bf16 with on-the-fly fp32->bf16 convert.
// The weight stream is non-temporal (TH=NT) so it cannot blow out the 192 MB
// L2 that holds the activations / intermediates; write-once outputs use NT
// stores for the same reason.
// ---------------------------------------------------------------------------

typedef __bf16 bf16;
typedef __attribute__((ext_vector_type(16))) __bf16 v16bf;
typedef __attribute__((ext_vector_type(8)))  __bf16 v8bf;
typedef __attribute__((ext_vector_type(8)))  float  v8f;
typedef __attribute__((ext_vector_type(4)))  float  v4f;

#define B_DIM   32
#define LQ      64
#define LK      64
#define DMODEL  1024
#define NHEAD   16
#define HD      64      // per-head dim (DK == DV == 64)

__device__ __forceinline__ v16bf cat16(v8bf lo, v8bf hi) {
  return __builtin_shufflevector(lo, hi, 0,1,2,3,4,5,6,7,8,9,10,11,12,13,14,15);
}
__device__ __forceinline__ v8bf ld8(const bf16* p) { return *(const v8bf*)p; }

// convert 8 consecutive f32 (16B-aligned) -> v8bf
__device__ __forceinline__ v8bf cvt8(const float* __restrict__ p) {
  float4 f0 = *(const float4*)p;
  float4 f1 = *(const float4*)(p + 4);
  v8bf r;
  r[0]=(bf16)f0.x; r[1]=(bf16)f0.y; r[2]=(bf16)f0.z; r[3]=(bf16)f0.w;
  r[4]=(bf16)f1.x; r[5]=(bf16)f1.y; r[6]=(bf16)f1.z; r[7]=(bf16)f1.w;
  return r;
}

__device__ __forceinline__ v8f wmma_bf16(v16bf a, v16bf b, v8f c) {
  // D = A(16x32) * B(32x16) + C(16x16 f32)
  return __builtin_amdgcn_wmma_f32_16x16x32_bf16(
      /*neg_a=*/false, a, /*neg_b=*/false, b,
      /*c_mod=*/(short)0, c, /*reuse_a=*/false, /*reuse_b=*/false);
}

// ---------------------------------------------------------------------------
// Kernel 1: GroupLinear projection.  For each position n (0..63):
//   Y[:, n, :] = X[:, n, :] (32x1024) @ W[n] (1024x1024), output bf16.
// grid (64 positions, 8 col-blocks of 128); block 256 = 8 waves.
// Each wave: 16 cols x both 16-row M-tiles, K-loop of 32.
// Weights stream once -> non-temporal loads; X tile is tiny + L2-resident.
// A-frag lane layout: row=lane%16, K = (lane/16)*16 + i.
// B-frag: col=lane%16, K = (lane/16)*16 + i.
// ---------------------------------------------------------------------------
__global__ __launch_bounds__(256)
void proj_kernel(const float* __restrict__ X, const float* __restrict__ W,
                 bf16* __restrict__ Y) {
  const int n    = blockIdx.x;
  const int cb   = blockIdx.y << 7;
  const int lane = threadIdx.x & 31;
  const int wv   = threadIdx.x >> 5;
  const int lx   = lane & 15;
  const int half = lane >> 4;
  const int col  = cb + (wv << 4) + lx;

  const float* Wn    = W + ((size_t)n << 20);               // n*1024*1024
  const float* Arow0 = X + (size_t)lx * (LQ * DMODEL) + (size_t)n * DMODEL;
  const float* Arow1 = Arow0 + (size_t)16 * (LQ * DMODEL);

  v8f acc0 = {}; v8f acc1 = {};
  for (int k0 = 0; k0 < DMODEL; k0 += 32) {
    const float* wp = Wn + (size_t)(k0 + (half << 4)) * DMODEL + col;
    if (k0 + 32 < DMODEL) __builtin_prefetch(wp + 32 * DMODEL, 0, 1);
    v16bf bfrag;
#pragma unroll
    for (int i = 0; i < 16; ++i)
      bfrag[i] = (bf16)__builtin_nontemporal_load(wp + (size_t)i * DMODEL);

    const int ka = k0 + (half << 4);
    v16bf a0 = cat16(cvt8(Arow0 + ka), cvt8(Arow0 + ka + 8));
    v16bf a1 = cat16(cvt8(Arow1 + ka), cvt8(Arow1 + ka + 8));
    acc0 = wmma_bf16(a0, bfrag, acc0);
    acc1 = wmma_bf16(a1, bfrag, acc1);
  }
  // C/D layout: vgpr j, lane -> (M = j + 8*half, N = lx)
  bf16* yp = Y + (size_t)n * DMODEL + col;
#pragma unroll
  for (int j = 0; j < 8; ++j) {
    const int m = j + (half << 3);
    yp[(size_t)m        * (LQ * DMODEL)] = (bf16)acc0[j];
    yp[(size_t)(m + 16) * (LQ * DMODEL)] = (bf16)acc1[j];
  }
}

// ---------------------------------------------------------------------------
// Kernel 2: attention for one (b, h) tile: S = QK^T/8, masked softmax,
// attn -> d_out (layout [H, B, Lq, Lk]),  ctx = P @ V -> bf16 workspace.
// block = 128 threads (4 waves); wave w owns q rows [16w, 16w+16).
// ---------------------------------------------------------------------------
__global__ __launch_bounds__(128)
void attn_kernel(const bf16* __restrict__ qp, const bf16* __restrict__ kp,
                 const bf16* __restrict__ vp, const int* __restrict__ mask,
                 bf16* __restrict__ ctx, float* __restrict__ attn_out) {
  __shared__ bf16 Vs[64 * 72];       // V transposed: [d][kpos], stride 72
  __shared__ bf16 Ps[4][16 * 72];    // per-wave P tile: [m][k], stride 72

  const int b    = blockIdx.x >> 4;
  const int h    = blockIdx.x & 15;
  const int tid  = threadIdx.x;
  const int lane = tid & 31;
  const int wv   = tid >> 5;
  const int lx   = lane & 15;
  const int half = lane >> 4;

  // ---- stage V transposed into LDS (coalesced 64B loads, b16 scatter) ----
  {
    const int r  = tid >> 1;             // kpos 0..63
    const int d0 = (tid & 1) << 5;       // 0 or 32
    const bf16* src = vp + (size_t)b * (LK * DMODEL) + (size_t)r * DMODEL
                         + h * HD + d0;
    v8bf x0 = ld8(src), x1 = ld8(src + 8), x2 = ld8(src + 16), x3 = ld8(src + 24);
#pragma unroll
    for (int j = 0; j < 8; ++j) {
      Vs[(d0 + j)      * 72 + r] = x0[j];
      Vs[(d0 + 8 + j)  * 72 + r] = x1[j];
      Vs[(d0 + 16 + j) * 72 + r] = x2[j];
      Vs[(d0 + 24 + j) * 72 + r] = x3[j];
    }
  }
  __syncthreads();

  const int qb = wv << 4;

  // ---- Q A-fragments (row-major qp, contiguous in d) ----
  v16bf aq[2];
#pragma unroll
  for (int ks = 0; ks < 2; ++ks) {
    const bf16* p = qp + (size_t)b * (LQ * DMODEL) + (size_t)(qb + lx) * DMODEL
                       + h * HD + ks * 32 + (half << 4);
    aq[ks] = cat16(ld8(p), ld8(p + 8));
  }

  // ---- S = Q K^T  (B-frag = contiguous d of one k-row => K^T for free) ----
  v8f s[4];
#pragma unroll
  for (int nt = 0; nt < 4; ++nt) {
    v8f acc = {};
#pragma unroll
    for (int ks = 0; ks < 2; ++ks) {
      const bf16* p = kp + (size_t)b * (LK * DMODEL)
                         + (size_t)((nt << 4) + lx) * DMODEL
                         + h * HD + ks * 32 + (half << 4);
      acc = wmma_bf16(aq[ks], cat16(ld8(p), ld8(p + 8)), acc);
    }
    s[nt] = acc;
  }

  // ---- scale + mask ----
  const int qrow = qb + (half << 3);
  const int* mrow = mask + (size_t)b * (LQ * LK);
#pragma unroll
  for (int nt = 0; nt < 4; ++nt)
#pragma unroll
    for (int j = 0; j < 8; ++j) {
      const int mv = mrow[(qrow + j) * LK + (nt << 4) + lx];
      s[nt][j] = (mv > 0) ? s[nt][j] * 0.125f : -1.0e9f;
    }

  // ---- row softmax: reduce across 16 lanes (N) and 4 N-tiles ----
  float rmax[8], rinv[8];
#pragma unroll
  for (int j = 0; j < 8; ++j) {
    float m = fmaxf(fmaxf(s[0][j], s[1][j]), fmaxf(s[2][j], s[3][j]));
    m = fmaxf(m, __shfl_xor(m, 1, 32));
    m = fmaxf(m, __shfl_xor(m, 2, 32));
    m = fmaxf(m, __shfl_xor(m, 4, 32));
    m = fmaxf(m, __shfl_xor(m, 8, 32));
    rmax[j] = m;
  }
#pragma unroll
  for (int nt = 0; nt < 4; ++nt)
#pragma unroll
    for (int j = 0; j < 8; ++j)
      s[nt][j] = __expf(s[nt][j] - rmax[j]);
#pragma unroll
  for (int j = 0; j < 8; ++j) {
    float t = (s[0][j] + s[1][j]) + (s[2][j] + s[3][j]);
    t += __shfl_xor(t, 1, 32);
    t += __shfl_xor(t, 2, 32);
    t += __shfl_xor(t, 4, 32);
    t += __shfl_xor(t, 8, 32);
    rinv[j] = 1.0f / t;
  }

  // ---- write attn (f32, write-once -> NT) + stage P (bf16) in LDS ----
  float* aout = attn_out + (size_t)(h * B_DIM + b) * (LQ * LK);
#pragma unroll
  for (int nt = 0; nt < 4; ++nt)
#pragma unroll
    for (int j = 0; j < 8; ++j) {
      const float p = s[nt][j] * rinv[j];
      __builtin_nontemporal_store(p, aout + (size_t)(qrow + j) * LK + (nt << 4) + lx);
      Ps[wv][((half << 3) + j) * 72 + (nt << 4) + lx] = (bf16)p;
    }
  // DS ops are in-order within a wave; Ps is wave-private -> no barrier.

  // ---- ctx = P @ V ----
  v16bf ap2[2];
#pragma unroll
  for (int ks = 0; ks < 2; ++ks) {
    const bf16* p = &Ps[wv][lx * 72 + ks * 32 + (half << 4)];
    ap2[ks] = cat16(ld8(p), ld8(p + 8));
  }
  bf16* cp = ctx + (size_t)b * (LQ * DMODEL) + h * HD;
#pragma unroll
  for (int dt = 0; dt < 4; ++dt) {
    v8f acc = {};
#pragma unroll
    for (int ks = 0; ks < 2; ++ks) {
      const bf16* p = &Vs[((dt << 4) + lx) * 72 + ks * 32 + (half << 4)];
      acc = wmma_bf16(ap2[ks], cat16(ld8(p), ld8(p + 8)), acc);
    }
#pragma unroll
    for (int j = 0; j < 8; ++j)
      cp[(size_t)(qrow + j) * DMODEL + (dt << 4) + lx] = (bf16)acc[j];
  }
}

// ---------------------------------------------------------------------------
// Kernel 3: out = ctx(2048x1024,bf16) @ fc_w^T + fc_b + residual(v) -> f32 ws.
// fc_w is [D, H*DV] row-major, so fc_w^T fragments are 64B-contiguous per
// lane; fc_w is 4 MB and re-read by 64 row-blocks -> keep regular-temporal.
// ---------------------------------------------------------------------------
__global__ __launch_bounds__(256)
void fc_kernel(const bf16* __restrict__ ctx, const float* __restrict__ fcw,
               const float* __restrict__ fcb, const float* __restrict__ vres,
               float* __restrict__ out) {
  const int mb   = blockIdx.x << 5;                 // 64 row blocks of 32
  const int lane = threadIdx.x & 31;
  const int wv   = threadIdx.x >> 5;
  const int lx   = lane & 15;
  const int half = lane >> 4;
  const int col  = (blockIdx.y << 7) + (wv << 4) + lx;

  const bf16*  a0p  = ctx + (size_t)(mb + lx) * DMODEL;
  const bf16*  a1p  = a0p + (size_t)16 * DMODEL;
  const float* wrow = fcw + (size_t)col * DMODEL;

  v8f acc0 = {}, acc1 = {};
  for (int k0 = 0; k0 < DMODEL; k0 += 32) {
    const float* wp = wrow + k0 + (half << 4);
    v16bf bfrag = cat16(cvt8(wp), cvt8(wp + 8));
    const int ka = k0 + (half << 4);
    v16bf af0 = cat16(ld8(a0p + ka), ld8(a0p + ka + 8));
    v16bf af1 = cat16(ld8(a1p + ka), ld8(a1p + ka + 8));
    acc0 = wmma_bf16(af0, bfrag, acc0);
    acc1 = wmma_bf16(af1, bfrag, acc1);
  }
  const float bias = fcb[col];
#pragma unroll
  for (int j = 0; j < 8; ++j) {
    const int m0 = mb + (half << 3) + j;
    out[(size_t)m0 * DMODEL + col] = acc0[j] + bias + vres[(size_t)m0 * DMODEL + col];
    const int m1 = m0 + 16;
    out[(size_t)m1 * DMODEL + col] = acc1[j] + bias + vres[(size_t)m1 * DMODEL + col];
  }
}

// ---------------------------------------------------------------------------
// Kernel 4: LayerNorm over D=1024, one row per block (256 threads x float4).
// Final output is write-once -> NT stores (native clang vector type).
// ---------------------------------------------------------------------------
__global__ __launch_bounds__(256)
void ln_kernel(const float* __restrict__ x, const float* __restrict__ g,
               const float* __restrict__ bt, float* __restrict__ out) {
  __shared__ float rs[8], rq[8];
  const int row  = blockIdx.x;
  const int t    = threadIdx.x;
  const int lane = t & 31;
  const int wv   = t >> 5;

  const float* xr = x + (size_t)row * DMODEL;
  v4f v = *(const v4f*)(xr + (t << 2));
  float s = (v[0] + v[1]) + (v[2] + v[3]);
  float q = (v[0] * v[0] + v[1] * v[1]) + (v[2] * v[2] + v[3] * v[3]);
#pragma unroll
  for (int d = 1; d < 32; d <<= 1) {
    s += __shfl_xor(s, d, 32);
    q += __shfl_xor(q, d, 32);
  }
  if (lane == 0) { rs[wv] = s; rq[wv] = q; }
  __syncthreads();
  float S = 0.f, Q = 0.f;
#pragma unroll
  for (int i = 0; i < 8; ++i) { S += rs[i]; Q += rq[i]; }
  const float mu   = S * (1.0f / DMODEL);
  const float var  = Q * (1.0f / DMODEL) - mu * mu;
  const float rstd = rsqrtf(var + 1e-5f);

  v4f gg = *(const v4f*)(g  + (t << 2));
  v4f bb = *(const v4f*)(bt + (t << 2));
  v4f o;
  o[0] = (v[0] - mu) * rstd * gg[0] + bb[0];
  o[1] = (v[1] - mu) * rstd * gg[1] + bb[1];
  o[2] = (v[2] - mu) * rstd * gg[2] + bb[2];
  o[3] = (v[3] - mu) * rstd * gg[3] + bb[3];
  __builtin_nontemporal_store(o, (v4f*)(out + (size_t)row * DMODEL + (t << 2)));
}

// ---------------------------------------------------------------------------
extern "C" void kernel_launch(void* const* d_in, const int* in_sizes, int n_in,
                              void* d_out, int out_size, void* d_ws, size_t ws_size,
                              hipStream_t stream) {
  (void)in_sizes; (void)n_in; (void)out_size; (void)ws_size;

  const float* q    = (const float*)d_in[0];
  const float* k    = (const float*)d_in[1];
  const float* v    = (const float*)d_in[2];
  const int*   mask = (const int*)  d_in[3];
  const float* w_qs = (const float*)d_in[4];
  const float* w_ks = (const float*)d_in[5];
  const float* w_vs = (const float*)d_in[6];
  const float* fc_w = (const float*)d_in[7];
  const float* fc_b = (const float*)d_in[8];
  const float* ln_g = (const float*)d_in[9];
  const float* ln_b = (const float*)d_in[10];

  float* out      = (float*)d_out;
  float* attn_out = out + (size_t)B_DIM * LQ * DMODEL;   // 2,097,152 floats in

  // workspace layout (24 MB total)
  char* ws = (char*)d_ws;
  bf16*  qp  = (bf16*)(ws);
  bf16*  kp  = (bf16*)(ws + ((size_t)4  << 20));
  bf16*  vp  = (bf16*)(ws + ((size_t)8  << 20));
  bf16*  ctx = (bf16*)(ws + ((size_t)12 << 20));
  float* fco = (float*)(ws + ((size_t)16 << 20));

  const dim3 gproj(64, 8);
  proj_kernel<<<gproj, 256, 0, stream>>>(q, w_qs, qp);
  proj_kernel<<<gproj, 256, 0, stream>>>(k, w_ks, kp);
  proj_kernel<<<gproj, 256, 0, stream>>>(v, w_vs, vp);

  attn_kernel<<<B_DIM * NHEAD, 128, 0, stream>>>(qp, kp, vp, mask, ctx, attn_out);

  fc_kernel<<<dim3(64, 8), 256, 0, stream>>>(ctx, fc_w, fc_b, v, fco);

  ln_kernel<<<B_DIM * LQ, 256, 0, stream>>>(fco, ln_g, ln_b, out);
}
